// DCNV4_73538430042723
// MI455X (gfx1250) — matrix operand order repaired and checked by reading, not compiled, not run on hardware
//
#include <hip/hip_runtime.h>

typedef __attribute__((ext_vector_type(16))) _Float16 v16h;
typedef __attribute__((ext_vector_type(8)))  _Float16 h8;
typedef __attribute__((ext_vector_type(8)))  float    v8f;

#define N_    4
#define C_    128
#define H_    128
#define W_    128
#define HW_   (H_ * W_)
#define CHW_  (C_ * HW_)
#define L_    (N_ * HW_)
#define G_    4
#define Cg_   32
#define OMD_  112        // OM_DIM (padded); 108 used

// ---------------------------------------------------------------------------
// WMMA helpers (gfx1250 wave32, 16x16x32 f16 -> f32)
// ---------------------------------------------------------------------------
static __device__ __forceinline__ v8f wmma_f16(v16h a, v16h b, v8f c) {
  // (neg_a, A, neg_b, B, c_mod, C, reuse_a, reuse_b)
  return __builtin_amdgcn_wmma_f32_16x16x32_f16(false, a, false, b, (short)0, c,
                                                false, false);
}

// B tile from LDS weight matrix stored row-major [out][in] as f16.
// B[k][n] = W[n][k]; per lane: K halves {8h..8h+7, 16+8h..16+8h+7} of column (nt*16+lane&15).
static __device__ __forceinline__ v16h load_B_lds(const _Float16* s_w, int nt,
                                                  int kk, int lane) {
  const int col = nt * 16 + (lane & 15);
  const int c0  = kk + ((lane >> 4) ? 8 : 0);
  const h8* p = (const h8*)(s_w + col * C_ + c0);
  h8 lo = p[0];
  h8 hi = p[2];
  return __builtin_shufflevector(lo, hi, 0, 1, 2, 3, 4, 5, 6, 7, 8, 9, 10, 11,
                                 12, 13, 14, 15);
}

// A tile from a row-major f16 [16][C_] LDS region (per-wave feat staging).
static __device__ __forceinline__ v16h load_A_lds(const _Float16* s_f, int kk,
                                                  int lane) {
  const int row = lane & 15;
  const int c0  = kk + ((lane >> 4) ? 8 : 0);
  const _Float16* rp = s_f + row * C_;
  h8 lo = *(const h8*)(rp + c0);
  h8 hi = *(const h8*)(rp + c0 + 16);
  return __builtin_shufflevector(lo, hi, 0, 1, 2, 3, 4, 5, 6, 7, 8, 9, 10, 11,
                                 12, 13, 14, 15);
}

// A tile gathered from x (NCHW): A[row][c] = x[n, c, hw0+row], f32 -> f16.
static __device__ __forceinline__ v16h load_A_x(const float* __restrict__ x,
                                                int n, int hw0, int kk,
                                                int lane) {
  const int row = lane & 15;
  const int c0  = kk + ((lane >> 4) ? 8 : 0);
  const float* base = x + (size_t)n * CHW_ + hw0 + row;
  v16h a;
#pragma unroll
  for (int j = 0; j < 8; ++j) {
    a[j]     = (_Float16)base[(size_t)(c0 + j) * HW_];
    a[j + 8] = (_Float16)base[(size_t)(c0 + 16 + j) * HW_];
  }
  return a;
}

// A tile from a row-major f16 [L][C_] global buffer.
static __device__ __forceinline__ v16h load_A_h16(
    const _Float16* __restrict__ y, int l0, int kk, int lane) {
  const int row = lane & 15;
  const int c0  = kk + ((lane >> 4) ? 8 : 0);
  const _Float16* rp = y + (size_t)(l0 + row) * C_;
  h8 lo = *(const h8*)(rp + c0);
  h8 hi = *(const h8*)(rp + c0 + 16);
  return __builtin_shufflevector(lo, hi, 0, 1, 2, 3, 4, 5, 6, 7, 8, 9, 10, 11,
                                 12, 13, 14, 15);
}

// ---------------------------------------------------------------------------
// Kernel 1: fused  feat = x*conv_w^T + b ;  value = feat*value_w^T + b (f16)
//                  om   = feat*om_w^T   + b (f32).  feat never leaves LDS.
// One wave per 16-row strip, 8 waves per block.
// ---------------------------------------------------------------------------
__global__ __launch_bounds__(256) void k1_fused_proj(
    const float* __restrict__ x, const float* __restrict__ conv_w,
    const float* __restrict__ conv_b, const float* __restrict__ value_w,
    const float* __restrict__ value_b, const float* __restrict__ om_w,
    const float* __restrict__ om_b, _Float16* __restrict__ valbuf,
    float* __restrict__ ombuf) {
  __shared__ _Float16 s_w[C_ * C_];        // 32 KB: reused for all 3 weights
  __shared__ _Float16 s_feat[8 * 16 * C_]; // 32 KB: per-wave feat staging

  const int tid   = threadIdx.x;
  const int lane  = tid & 31;
  const int wid   = tid >> 5;
  const int strip = blockIdx.x * 8 + wid;
  const int l0    = strip * 16;
  const int n     = l0 / HW_;
  const int hw0   = l0 % HW_;

  const int colr = lane & 15;
  const int rb   = (lane >> 4) * 8;

  // phase 0: conv_w -> LDS f16
  for (int i = tid; i < C_ * C_; i += 256) s_w[i] = (_Float16)conv_w[i];
  __syncthreads();

  // A tiles of x (kept in registers for all 8 N tiles)
  v16h a0 = load_A_x(x, n, hw0, 0, lane);
  v16h a1 = load_A_x(x, n, hw0, 32, lane);
  v16h a2 = load_A_x(x, n, hw0, 64, lane);
  v16h a3 = load_A_x(x, n, hw0, 96, lane);

  _Float16* s_f = s_feat + wid * 16 * C_;
#pragma unroll
  for (int nt = 0; nt < 8; ++nt) {
    v8f acc = {};
    acc = wmma_f16(a0, load_B_lds(s_w, nt, 0, lane), acc);
    acc = wmma_f16(a1, load_B_lds(s_w, nt, 32, lane), acc);
    acc = wmma_f16(a2, load_B_lds(s_w, nt, 64, lane), acc);
    acc = wmma_f16(a3, load_B_lds(s_w, nt, 96, lane), acc);
    const int col   = nt * 16 + colr;
    const float bia = conv_b[col];
#pragma unroll
    for (int i = 0; i < 8; ++i)
      s_f[(rb + i) * C_ + col] = (_Float16)(acc[i] + bia);
  }
  __syncthreads();

  // phase 1: value_w -> LDS
  for (int i = tid; i < C_ * C_; i += 256) s_w[i] = (_Float16)value_w[i];
  __syncthreads();

  v16h f0 = load_A_lds(s_f, 0, lane);
  v16h f1 = load_A_lds(s_f, 32, lane);
  v16h f2 = load_A_lds(s_f, 64, lane);
  v16h f3 = load_A_lds(s_f, 96, lane);

#pragma unroll
  for (int nt = 0; nt < 8; ++nt) {
    v8f acc = {};
    acc = wmma_f16(f0, load_B_lds(s_w, nt, 0, lane), acc);
    acc = wmma_f16(f1, load_B_lds(s_w, nt, 32, lane), acc);
    acc = wmma_f16(f2, load_B_lds(s_w, nt, 64, lane), acc);
    acc = wmma_f16(f3, load_B_lds(s_w, nt, 96, lane), acc);
    const int col   = nt * 16 + colr;
    const float bia = value_b[col];
#pragma unroll
    for (int i = 0; i < 8; ++i)
      valbuf[(size_t)(l0 + rb + i) * C_ + col] = (_Float16)(acc[i] + bia);
  }
  __syncthreads();

  // phase 2: om_w (112 rows) -> LDS
  for (int i = tid; i < OMD_ * C_; i += 256) s_w[i] = (_Float16)om_w[i];
  __syncthreads();

#pragma unroll
  for (int nt = 0; nt < 7; ++nt) {
    v8f acc = {};
    acc = wmma_f16(f0, load_B_lds(s_w, nt, 0, lane), acc);
    acc = wmma_f16(f1, load_B_lds(s_w, nt, 32, lane), acc);
    acc = wmma_f16(f2, load_B_lds(s_w, nt, 64, lane), acc);
    acc = wmma_f16(f3, load_B_lds(s_w, nt, 96, lane), acc);
    const int col   = nt * 16 + colr;
    const float bia = om_b[col];
#pragma unroll
    for (int i = 0; i < 8; ++i)
      ombuf[(size_t)(l0 + rb + i) * OMD_ + col] = acc[i] + bia;
  }
}

// ---------------------------------------------------------------------------
// Kernel 2: deformable bilinear gather. One wave per (l, g); lane = channel.
// Each tap is a coalesced 64B f16 read (value stays L2-resident, 192MB L2).
// ---------------------------------------------------------------------------
__global__ __launch_bounds__(256) void k2_dcn(
    const _Float16* __restrict__ valbuf, const float* __restrict__ ombuf,
    _Float16* __restrict__ ybuf) {
  const int lane = threadIdx.x & 31;
  const int wid  = threadIdx.x >> 5;
  const int lg   = blockIdx.x * 8 + wid;
  const int l    = lg >> 2;
  const int g    = lg & 3;
  const int n    = l >> 14;
  const int hw   = l & (HW_ - 1);
  const int h    = hw >> 7;
  const int w    = hw & (W_ - 1);

  const float* om = ombuf + (size_t)l * OMD_ + g * 27;
  const _Float16* vb =
      valbuf + (size_t)n * HW_ * C_ + g * Cg_ + lane;

  float acc = 0.f;
#pragma unroll
  for (int k = 0; k < 9; ++k) {
    const float px = (float)(w + (k % 3) - 1) + om[2 * k];
    const float py = (float)(h + (k / 3) - 1) + om[2 * k + 1];
    const float mk = om[18 + k];
    const float fx = floorf(px), fy = floorf(py);
    const float lx = px - fx, ly = py - fy;
    const int x0 = (int)fx, y0 = (int)fy;
    float s = 0.f;
#pragma unroll
    for (int t = 0; t < 4; ++t) {
      const int xi = x0 + (t & 1);
      const int yi = y0 + (t >> 1);
      const float wx = (t & 1) ? lx : 1.f - lx;
      const float wy = (t >> 1) ? ly : 1.f - ly;
      const bool ok = (xi >= 0) && (xi < W_) && (yi >= 0) && (yi < H_);
      const int xc = xi < 0 ? 0 : (xi > W_ - 1 ? W_ - 1 : xi);
      const int yc = yi < 0 ? 0 : (yi > H_ - 1 ? H_ - 1 : yi);
      const float v = (float)vb[(size_t)(yc * W_ + xc) * C_];
      s += ok ? (wx * wy * v) : 0.f;
    }
    acc += mk * s;
  }
  ybuf[(size_t)l * C_ + g * Cg_ + lane] = (_Float16)acc;
}

// ---------------------------------------------------------------------------
// Kernel 3: out = y*out_w^T + out_b, written to NCHW with an LDS transpose so
// stores are contiguous along hw.
// ---------------------------------------------------------------------------
__global__ __launch_bounds__(256) void k3_out(
    const _Float16* __restrict__ ybuf, const float* __restrict__ out_w,
    const float* __restrict__ out_b, float* __restrict__ out) {
  __shared__ _Float16 s_w[C_ * C_];   // 32 KB
  __shared__ float s_t[8 * 16 * 16];  // 8 KB: per-wave 16x16 transpose tile

  const int tid   = threadIdx.x;
  const int lane  = tid & 31;
  const int wid   = tid >> 5;
  const int strip = blockIdx.x * 8 + wid;
  const int l0    = strip * 16;
  const int n     = l0 / HW_;
  const int hw0   = l0 % HW_;

  for (int i = tid; i < C_ * C_; i += 256) s_w[i] = (_Float16)out_w[i];
  __syncthreads();

  v16h a0 = load_A_h16(ybuf, l0, 0, lane);
  v16h a1 = load_A_h16(ybuf, l0, 32, lane);
  v16h a2 = load_A_h16(ybuf, l0, 64, lane);
  v16h a3 = load_A_h16(ybuf, l0, 96, lane);

  float* st = s_t + wid * 256;
  const int colr = lane & 15;
  const int rb   = (lane >> 4) * 8;
#pragma unroll
  for (int nt = 0; nt < 8; ++nt) {
    v8f acc = {};
    acc = wmma_f16(a0, load_B_lds(s_w, nt, 0, lane), acc);
    acc = wmma_f16(a1, load_B_lds(s_w, nt, 32, lane), acc);
    acc = wmma_f16(a2, load_B_lds(s_w, nt, 64, lane), acc);
    acc = wmma_f16(a3, load_B_lds(s_w, nt, 96, lane), acc);
    const int col   = nt * 16 + colr;
    const float bia = out_b[col];
    // stage [col_local][row] so the readback is contiguous along rows
#pragma unroll
    for (int i = 0; i < 8; ++i) st[colr * 16 + rb + i] = acc[i] + bia;
    __syncthreads();
#pragma unroll
    for (int it = 0; it < 8; ++it) {
      const int oc = 2 * it + (lane >> 4);  // 0..15
      const int r  = lane & 15;
      out[(size_t)n * CHW_ + (size_t)(nt * 16 + oc) * HW_ + hw0 + r] =
          st[oc * 16 + r];
    }
    __syncthreads();
  }
}

// ---------------------------------------------------------------------------
extern "C" void kernel_launch(void* const* d_in, const int* in_sizes, int n_in,
                              void* d_out, int out_size, void* d_ws,
                              size_t ws_size, hipStream_t stream) {
  (void)in_sizes; (void)n_in; (void)out_size; (void)ws_size;
  const float* x       = (const float*)d_in[0];
  const float* conv_w  = (const float*)d_in[1];
  const float* conv_b  = (const float*)d_in[2];
  const float* value_w = (const float*)d_in[3];
  const float* value_b = (const float*)d_in[4];
  const float* om_w    = (const float*)d_in[5];
  const float* om_b    = (const float*)d_in[6];
  const float* out_w   = (const float*)d_in[7];
  const float* out_b   = (const float*)d_in[8];
  float* out = (float*)d_out;

  // workspace: value f16 (16 MiB) | om f32 (28 MiB) | y f16 (16 MiB) = 60 MiB
  _Float16* valbuf = (_Float16*)d_ws;
  float* ombuf = (float*)((char*)d_ws + (size_t)L_ * C_ * sizeof(_Float16));
  _Float16* ybuf =
      (_Float16*)((char*)d_ws + (size_t)L_ * C_ * sizeof(_Float16) +
                  (size_t)L_ * OMD_ * sizeof(float));

  k1_fused_proj<<<L_ / 16 / 8, 256, 0, stream>>>(
      x, conv_w, conv_b, value_w, value_b, om_w, om_b, valbuf, ombuf);
  k2_dcn<<<L_ * G_ / 8, 256, 0, stream>>>(valbuf, ombuf, ybuf);
  k3_out<<<L_ / 16 / 8, 256, 0, stream>>>(ybuf, out_w, out_b, out);
}
